// NeRF_Instant_63677185131090
// MI455X (gfx1250) — compile-verified, matrix operand order checked by reading, not run on hardware
//
#include <hip/hip_runtime.h>
#include <hip/hip_bf16.h>
#include <math.h>

typedef __attribute__((ext_vector_type(16))) _Float16 v16h;
typedef __attribute__((ext_vector_type(8)))  float    v8f;

#define WMMA_F16(A_, B_, C_) \
  __builtin_amdgcn_wmma_f32_16x16x32_f16(false, (A_), false, (B_), (short)0, (C_), false, false)

#define HASH_T (1u << 19)
#define HASH_MASK (HASH_T - 1u)

// Swizzled operand-tile layout: a 32(K) x 16(row/col) f16 WMMA operand is stored
// as [lane 0..31][half 0..15] so each lane reads its 16 halves contiguously
// (32 bytes -> 2 x ds_load_b128). Derived from ISA 7.12.2 16-bit A/B striping:
//   lane = rc + 16*((kk>>3)&1),  half = (kk&7) + ((kk&16)>>1)
__device__ __forceinline__ int swz(int rc, int kk) {
  return (((rc & 15) + ((kk & 8) << 1)) << 4) + (kk & 7) + ((kk & 16) >> 1);
}
// activation element (row m, feature k in [0,64)) inside a [2][32][16]-half buffer
__device__ __forceinline__ int aidx(int m, int k) {
  return ((k >> 5) << 9) + swz(m, k & 31);
}

__global__ __launch_bounds__(256) void nerf_instant_fused(
    const float* __restrict__ x, const float* __restrict__ dirs,
    const float* __restrict__ tables,
    const float* __restrict__ gW1d, const float* __restrict__ gb1d,
    const float* __restrict__ gW2d, const float* __restrict__ gb2d,
    const float* __restrict__ gW1c, const float* __restrict__ gb1c,
    const float* __restrict__ gW2c, const float* __restrict__ gb2c,
    const float* __restrict__ gW3c, const float* __restrict__ gb3c,
    float* __restrict__ out, int n) {
  // ---- LDS (~57 KB): everything pre-swizzled into WMMA operand layout ----
  // weight operand tiles (32K x 16N each, 1 KB each):
  //  0..3  W1d  (kc0, nt0..3)         4..5  W2d  (kc0,kc1)
  //  6..13 W1c  (nt*2+kc)            14..21 W2c  (nt*2+kc)       22..23 W3c (kc)
  __shared__ v16h wtiles[24][32];
  __shared__ v16h featsw[16][32];     // 16 point-tiles x (16 pts x 32 feats)
  __shared__ v16h actsw[8][2][32];    // per-wave 16 x 64 activation tile
  __shared__ float sb1d[64], sb2d[16], sb1c[64], sb2c[64], sb3c[16];

  const int tid  = threadIdx.x;
  const int lane = tid & 31;
  const int wv   = tid >> 5;
  const int m    = lane & 15;   // row (A) / column (B/D) owned by this lane
  const int hi   = lane >> 4;

  _Float16* wt = (_Float16*)wtiles;

  // ---- stage weights f32 -> f16, swizzled, zero-padded ----
  for (int i = tid; i < 32 * 64; i += 256) {       // W1d [32][64]
    int k = i >> 6, c = i & 63;
    wt[(c >> 4) * 512 + swz(c & 15, k)] = (_Float16)gW1d[i];
  }
  for (int i = tid; i < 64 * 16; i += 256) {       // W2d [64][16]
    int k = i >> 4, c = i & 15;
    wt[(4 + (k >> 5)) * 512 + swz(c, k & 31)] = (_Float16)gW2d[i];
  }
  for (int i = tid; i < 64 * 64; i += 256) {       // W1c [42->64][64]
    int k = i >> 6, c = i & 63;
    float v = (k < 42) ? gW1c[k * 64 + c] : 0.0f;
    wt[(6 + ((c >> 4) << 1) + (k >> 5)) * 512 + swz(c & 15, k & 31)] = (_Float16)v;
  }
  for (int i = tid; i < 64 * 64; i += 256) {       // W2c [64][64]
    int k = i >> 6, c = i & 63;
    wt[(14 + ((c >> 4) << 1) + (k >> 5)) * 512 + swz(c & 15, k & 31)] = (_Float16)gW2c[i];
  }
  for (int i = tid; i < 64 * 16; i += 256) {       // W3c [64][3->16]
    int k = i >> 4, c = i & 15;
    float v = (c < 3) ? gW3c[k * 3 + c] : 0.0f;
    wt[(22 + (k >> 5)) * 512 + swz(c, k & 31)] = (_Float16)v;
  }
  if (tid < 64) { sb1d[tid] = gb1d[tid]; sb1c[tid] = gb1c[tid]; sb2c[tid] = gb2c[tid]; }
  if (tid < 16) { sb2d[tid] = gb2d[tid]; sb3c[tid] = gb3c[tid]; }

  // ---- phase 1: hash-grid encode, one point per thread ----
  const int p = blockIdx.x * 256 + tid;
  const bool valid = p < n;
  const int pc = valid ? p : 0;
  float xx = x[pc * 3 + 0], xy = x[pc * 3 + 1], xz = x[pc * 3 + 2];
  xx = fminf(fmaxf(xx, -2.0f), 2.0f - 1e-6f);
  xy = fminf(fmaxf(xy, -2.0f), 2.0f - 1e-6f);
  xz = fminf(fmaxf(xz, -2.0f), 2.0f - 1e-6f);

  _Float16* fs = (_Float16*)featsw[tid >> 4];  // this thread's point-tile
  const int frow = tid & 15;

  const float RES[16] = {16.f, 20.f, 25.f, 32.f, 40.f, 50.f, 64.f, 80.f,
                         101.f, 128.f, 161.f, 203.f, 256.f, 322.f, 406.f, 512.f};
#pragma unroll
  for (int l = 0; l < 16; ++l) {
    const float inv_grid = RES[l] * 0.25f;  // 1/grid, grid = 4/res
    const float fx = (xx + 2.0f) * inv_grid;
    const float fy = (xy + 2.0f) * inv_grid;
    const float fz = (xz + 2.0f) * inv_grid;
    const int bx = (int)floorf(fx), by = (int)floorf(fy), bz = (int)floorf(fz);
    const float wx = fx - (float)bx, wy = fy - (float)by, wz = fz - (float)bz;
    const float* tl = tables + (size_t)l * (size_t)(HASH_T * 2);
    float f0 = 0.f, f1 = 0.f;
#pragma unroll
    for (int c = 0; c < 8; ++c) {
      const int oi = (c >> 2) & 1, oj = (c >> 1) & 1, ok = c & 1;
      unsigned h = (unsigned)(bx + oi) * 1u ^
                   (unsigned)(by + oj) * 2654435761u ^
                   (unsigned)(bz + ok) * 805459861u;
      const unsigned idx = h & HASH_MASK;
      const float2 v = *(const float2*)(tl + (size_t)idx * 2);  // L2-resident gather
      const float w = (oi ? wx : 1.f - wx) * (oj ? wy : 1.f - wy) * (ok ? wz : 1.f - wz);
      f0 = fmaf(w, v.x, f0);
      f1 = fmaf(w, v.y, f1);
    }
    fs[swz(frow, 2 * l + 0)] = (_Float16)(valid ? f0 : 0.f);
    fs[swz(frow, 2 * l + 1)] = (_Float16)(valid ? f1 : 0.f);
  }

  __syncthreads();

  // ---- phase 2: fused MLP, one 16-point tile per wave iteration ----
  _Float16* A = (_Float16*)actsw[wv];

#pragma unroll 1
  for (int tile = 0; tile < 2; ++tile) {
    const int ptile = wv * 2 + tile;
    const int pbase = blockIdx.x * 256 + ptile * 16;

    // L1 density: feat[16x32] @ W1d[32x64] + b1d, relu
    {
      const v16h a = featsw[ptile][lane];
#pragma unroll
      for (int nt = 0; nt < 4; ++nt) {
        const v16h b = wtiles[nt][lane];
        v8f c = {};
        c = WMMA_F16(a, b, c);
        const float bias = sb1d[nt * 16 + m];
#pragma unroll
        for (int v = 0; v < 8; ++v) {
          const float t = c[v] + bias;
          A[aidx(v + hi * 8, nt * 16 + m)] = (_Float16)(t > 0.f ? t : 0.f);
        }
      }
    }

    // L2 density: a1[16x64] @ W2d[64x16] + b2d  ->  sigma | geo
    {
      const v16h a0 = actsw[wv][0][lane];
      const v16h a1 = actsw[wv][1][lane];
      v8f c = {};
      c = WMMA_F16(a0, wtiles[4][lane], c);
      c = WMMA_F16(a1, wtiles[5][lane], c);
      const float bias = sb2d[m];
      if (m == 0) {  // column 0 -> sigma = softplus
#pragma unroll
        for (int v = 0; v < 8; ++v) {
          const int pt = pbase + v + hi * 8;
          if (pt < n) {
            const float s = c[v] + bias;
            out[(size_t)pt * 4 + 3] = (s > 20.f) ? s : log1pf(expf(s));
          }
        }
      } else {       // columns 1..15 -> geo into z cols 0..14
#pragma unroll
        for (int v = 0; v < 8; ++v)
          A[aidx(v + hi * 8, m - 1)] = (_Float16)(c[v] + bias);
      }
    }

    // direction encoding into z cols 15..41, zero-pad cols 42..63
    for (int e = lane; e < 16 * 49; e += 32) {
      const int rr = e / 49;
      const int cc = 15 + (e % 49);
      float val = 0.f;
      if (cc < 42) {
        const int j = cc - 15;
        const int pt = pbase + rr;
        const float* dp = dirs + (size_t)(pt < n ? pt : 0) * 3;
        if (j < 3) {
          val = dp[j];
        } else if (j < 15) {
          const int q = j - 3;
          val = __sinf((float)(1 << (q / 3)) * dp[q % 3]);
        } else {
          const int q = j - 15;
          val = __cosf((float)(1 << (q / 3)) * dp[q % 3]);
        }
      }
      A[aidx(rr, cc)] = (_Float16)val;
    }

    // L1 color: z[16x64(42)] @ W1c + b1c, relu
    {
      const v16h a0 = actsw[wv][0][lane];
      const v16h a1 = actsw[wv][1][lane];
#pragma unroll
      for (int nt = 0; nt < 4; ++nt) {
        v8f c = {};
        c = WMMA_F16(a0, wtiles[6 + nt * 2 + 0][lane], c);
        c = WMMA_F16(a1, wtiles[6 + nt * 2 + 1][lane], c);
        const float bias = sb1c[nt * 16 + m];
#pragma unroll
        for (int v = 0; v < 8; ++v) {
          const float t = c[v] + bias;
          A[aidx(v + hi * 8, nt * 16 + m)] = (_Float16)(t > 0.f ? t : 0.f);
        }
      }
    }

    // L2 color: [16x64] @ W2c + b2c, relu
    {
      const v16h a0 = actsw[wv][0][lane];
      const v16h a1 = actsw[wv][1][lane];
#pragma unroll
      for (int nt = 0; nt < 4; ++nt) {
        v8f c = {};
        c = WMMA_F16(a0, wtiles[14 + nt * 2 + 0][lane], c);
        c = WMMA_F16(a1, wtiles[14 + nt * 2 + 1][lane], c);
        const float bias = sb2c[nt * 16 + m];
#pragma unroll
        for (int v = 0; v < 8; ++v) {
          const float t = c[v] + bias;
          A[aidx(v + hi * 8, nt * 16 + m)] = (_Float16)(t > 0.f ? t : 0.f);
        }
      }
    }

    // L3 color: [16x64] @ W3c(pad 64x16) + b3c, sigmoid -> out cols 0..2
    {
      const v16h a0 = actsw[wv][0][lane];
      const v16h a1 = actsw[wv][1][lane];
      v8f c = {};
      c = WMMA_F16(a0, wtiles[22][lane], c);
      c = WMMA_F16(a1, wtiles[23][lane], c);
      if (m < 3) {
        const float bias = sb3c[m];
#pragma unroll
        for (int v = 0; v < 8; ++v) {
          const int pt = pbase + v + hi * 8;
          if (pt < n) {
            const float t = c[v] + bias;
            out[(size_t)pt * 4 + m] = 1.f / (1.f + expf(-t));
          }
        }
      }
    }
  }
}

extern "C" void kernel_launch(void* const* d_in, const int* in_sizes, int n_in,
                              void* d_out, int out_size, void* d_ws, size_t ws_size,
                              hipStream_t stream) {
  const float* x    = (const float*)d_in[0];
  const float* d    = (const float*)d_in[1];
  const float* tab  = (const float*)d_in[2];
  const float* W1d  = (const float*)d_in[3];
  const float* b1d  = (const float*)d_in[4];
  const float* W2d  = (const float*)d_in[5];
  const float* b2d  = (const float*)d_in[6];
  const float* W1c  = (const float*)d_in[7];
  const float* b1c  = (const float*)d_in[8];
  const float* W2c  = (const float*)d_in[9];
  const float* b2c  = (const float*)d_in[10];
  const float* W3c  = (const float*)d_in[11];
  const float* b3c  = (const float*)d_in[12];

  const int n = in_sizes[0] / 3;
  const int blocks = (n + 255) / 256;
  nerf_instant_fused<<<blocks, 256, 0, stream>>>(
      x, d, tab, W1d, b1d, W2d, b2d, W1c, b1c, W2c, b2c, W3c, b3c,
      (float*)d_out, n);
}
